// PointEncoder_27118423506988
// MI455X (gfx1250) — compile-verified, matrix operand order checked by reading, not run on hardware
//
#include <hip/hip_runtime.h>

// ---------------------------------------------------------------------------
// Point-MAE style encoder for MI455X (gfx1250, wave32, WMMA).
// B=16, N=8192, G=512, K=32, D=384, L=12, H=6, hd=64
// All GEMM B-operands are stored transposed (N x K row-major) so both LDS
// tiles stage with contiguous b128 copies and fragments are 2x ds_load_b128.
// LDS staging uses CDNA5 async global->LDS copies (ASYNCcnt).
// ---------------------------------------------------------------------------

#define USE_ASYNC 1

typedef __attribute__((ext_vector_type(16))) _Float16 v16h;
typedef __attribute__((ext_vector_type(8)))  _Float16 v8h;
typedef __attribute__((ext_vector_type(8)))  float    v8f;

#define Bb_ 16
#define Nn_ 8192
#define Gg_ 512
#define Kk_ 32
#define Dd_ 384
#define Ll_ 12
#define Hh_ 6
#define HD_ 64
#define BGK_ 262144LL   // B*G*K
#define BG_  8192LL     // B*G

// ------------------------------- reductions --------------------------------
__device__ inline float wave_sum(float v) {
#pragma unroll
  for (int o = 16; o > 0; o >>= 1) v += __shfl_down(v, o);
  return v;
}
__device__ inline float wave_max(float v) {
#pragma unroll
  for (int o = 16; o > 0; o >>= 1) v = fmaxf(v, __shfl_down(v, o));
  return v;
}

// ------------------------------- FPS ---------------------------------------
__global__ __launch_bounds__(1024)
void fps_kernel(const float* __restrict__ pts, float* __restrict__ center) {
  const int b = blockIdx.x, t = threadIdx.x;
  const float* P = pts + (long long)b * Nn_ * 3;
  float px[8], py[8], pz[8], dist[8];
#pragma unroll
  for (int i = 0; i < 8; ++i) {
    int p = t * 8 + i;
    px[i] = P[p * 3 + 0]; py[i] = P[p * 3 + 1]; pz[i] = P[p * 3 + 2];
    dist[i] = 1e10f;
  }
  __shared__ float sv[1024];
  __shared__ int   si[1024];
  __shared__ float cur[3];
  if (t == 0) {
    cur[0] = P[0]; cur[1] = P[1]; cur[2] = P[2];
    center[(long long)b * Gg_ * 3 + 0] = P[0];
    center[(long long)b * Gg_ * 3 + 1] = P[1];
    center[(long long)b * Gg_ * 3 + 2] = P[2];
  }
  __syncthreads();
  for (int it = 1; it < Gg_; ++it) {
    const float cx = cur[0], cy = cur[1], cz = cur[2];
    float best = -1.0f; int bi = 0;
#pragma unroll
    for (int i = 0; i < 8; ++i) {
      float dx = px[i] - cx, dy = py[i] - cy, dz = pz[i] - cz;
      float d = dx * dx + dy * dy + dz * dz;
      dist[i] = fminf(dist[i], d);
      if (dist[i] > best) { best = dist[i]; bi = t * 8 + i; }
    }
    sv[t] = best; si[t] = bi;
    __syncthreads();
    for (int s = 512; s > 0; s >>= 1) {
      if (t < s) {
        if (sv[t + s] > sv[t] || (sv[t + s] == sv[t] && si[t + s] < si[t])) {
          sv[t] = sv[t + s]; si[t] = si[t + s];
        }
      }
      __syncthreads();
    }
    if (t == 0) {
      int sel = si[0];
      float x = P[sel * 3], y = P[sel * 3 + 1], z = P[sel * 3 + 2];
      cur[0] = x; cur[1] = y; cur[2] = z;
      long long o = ((long long)b * Gg_ + it) * 3;
      center[o] = x; center[o + 1] = y; center[o + 2] = z;
    }
    __syncthreads();
  }
}

// ------------------------------- KNN + gather ------------------------------
__global__ __launch_bounds__(256)
void knn_kernel(const float* __restrict__ pts, const float* __restrict__ center,
                float* __restrict__ nbr) {
  const int bg = blockIdx.x, b = bg >> 9, t = threadIdx.x;
  const float* P = pts + (long long)b * Nn_ * 3;
  __shared__ float dist[Nn_];
  __shared__ float rv[256];
  __shared__ int   ri[256];
  __shared__ int   sel[Kk_];
  const float cx = center[bg * 3 + 0], cy = center[bg * 3 + 1], cz = center[bg * 3 + 2];
  for (int p = t; p < Nn_; p += 256) {
    float dx = P[p * 3] - cx, dy = P[p * 3 + 1] - cy, dz = P[p * 3 + 2] - cz;
    dist[p] = dx * dx + dy * dy + dz * dz;
  }
  __syncthreads();
  for (int k = 0; k < Kk_; ++k) {
    float best = 1e30f; int bi = 0;
    for (int p = t; p < Nn_; p += 256) {
      float d = dist[p];
      if (d < best || (d == best && p < bi)) { best = d; bi = p; }
    }
    rv[t] = best; ri[t] = bi;
    __syncthreads();
    for (int s = 128; s > 0; s >>= 1) {
      if (t < s && (rv[t + s] < rv[t] || (rv[t + s] == rv[t] && ri[t + s] < ri[t]))) {
        rv[t] = rv[t + s]; ri[t] = ri[t + s];
      }
      __syncthreads();
    }
    if (t == 0) { sel[k] = ri[0]; dist[ri[0]] = 1e30f; }
    __syncthreads();
  }
  if (t < Kk_ * 3) {
    int k = t / 3, c = t % 3;
    int p = sel[k];
    nbr[((long long)bg * Kk_ + k) * 3 + c] = P[p * 3 + c] - center[bg * 3 + c];
  }
}

// ----------------------- generic batched WMMA GEMM -------------------------
// C[M,N] = act(A[M,Kd] @ B[Kd,N] + bias) (+ resid), batched over z.
// B is given TRANSPOSED: BT[N][Kd] row-major.
// If SPLIT: A columns [0,splitK) come from A2[m/a2div][:] (group concat).
// Block = 128 thr = 4 waves; block tile 64x64; K-step 32.
// Fragment layout (CDNA5 16-bit): lane l (idx=l&15, koff=(l>>4)*8):
//   halves 0..7 = K[koff..koff+7], halves 8..15 = K[koff+16..koff+23].
template <bool SPLIT>
__global__ __launch_bounds__(128)
void wmma_gemm(const _Float16* __restrict__ A, long long sA,
               const _Float16* __restrict__ A2, int splitK, int a2div,
               const _Float16* __restrict__ BT, long long sB,
               const float* __restrict__ bias,
               const float* __restrict__ resid, long long sR,
               float* __restrict__ Cf, _Float16* __restrict__ Ch, long long sC,
               int M, int N, int Kd, int act) {
  const int bz = blockIdx.z;
  const _Float16* Ab = A + (long long)bz * sA;
  const _Float16* BTb = BT + (long long)bz * sB;
  const int m0 = blockIdx.y * 64, n0 = blockIdx.x * 64;
  const int tid = threadIdx.x, lane = tid & 31, wv = tid >> 5;
  const int ldA = Kd - (SPLIT ? splitK : 0);

  __shared__ __align__(16) _Float16 As[64][40];
  __shared__ __align__(16) _Float16 Bt[64][40];   // Bt[n][k]

  v8f acc[4];
#pragma unroll
  for (int j = 0; j < 4; ++j)
#pragma unroll
    for (int r = 0; r < 8; ++r) acc[j][r] = 0.0f;

  for (int k0 = 0; k0 < Kd; k0 += 32) {
#pragma unroll
    for (int r = 0; r < 2; ++r) {
      const int v = tid + r * 128;          // 256 vectors of 8 halves per tile
      const int row = v >> 2, c8 = (v & 3) * 8;
      const _Float16* gA;
      if (SPLIT && k0 < splitK)
        gA = A2 + (long long)((m0 + row) / a2div) * splitK + k0 + c8;
      else
        gA = Ab + (long long)(m0 + row) * ldA + (k0 - (SPLIT ? splitK : 0)) + c8;
      const _Float16* gB = BTb + (long long)(n0 + row) * Kd + k0 + c8;
#if USE_ASYNC
      const unsigned la = (unsigned)(size_t)&As[row][c8];
      const unsigned lb = (unsigned)(size_t)&Bt[row][c8];
      asm volatile("global_load_async_to_lds_b128 %0, %1, off"
                   :: "v"(la), "v"(gA) : "memory");
      asm volatile("global_load_async_to_lds_b128 %0, %1, off"
                   :: "v"(lb), "v"(gB) : "memory");
#else
      *(v8h*)&As[row][c8] = *(const v8h*)gA;
      *(v8h*)&Bt[row][c8] = *(const v8h*)gB;
#endif
    }
#if USE_ASYNC
    asm volatile("s_wait_asynccnt 0x0" ::: "memory");
#endif
    __syncthreads();

    const int mr = lane & 15, koff = (lane >> 4) * 8;
    v8h alo = *(const v8h*)&As[wv * 16 + mr][koff];
    v8h ahi = *(const v8h*)&As[wv * 16 + mr][koff + 16];
    v16h af;
#pragma unroll
    for (int i = 0; i < 8; ++i) { af[i] = alo[i]; af[i + 8] = ahi[i]; }

#pragma unroll
    for (int j = 0; j < 4; ++j) {
      v8h blo = *(const v8h*)&Bt[j * 16 + mr][koff];
      v8h bhi = *(const v8h*)&Bt[j * 16 + mr][koff + 16];
      v16h bf;
#pragma unroll
      for (int i = 0; i < 8; ++i) { bf[i] = blo[i]; bf[i + 8] = bhi[i]; }
      acc[j] = __builtin_amdgcn_wmma_f32_16x16x32_f16(
          false, af, false, bf, (short)0, acc[j], false, false);
    }
    __syncthreads();
  }

  // epilogue: D layout: elem r @ lane l -> m = (l>>4)*8 + r, n = l&15
  const int mr = lane & 15, rbase = (lane >> 4) * 8;
#pragma unroll
  for (int j = 0; j < 4; ++j) {
    const int gn = n0 + j * 16 + mr;
    const float bv = bias ? bias[gn] : 0.0f;
#pragma unroll
    for (int r = 0; r < 8; ++r) {
      const int gm = m0 + wv * 16 + rbase + r;
      float val = acc[j][r] + bv;
      if (act == 1) val = fmaxf(val, 0.0f);
      else if (act == 2) val = 0.5f * val * (1.0f + erff(val * 0.70710678118f));
      const long long ci = (long long)bz * sC + (long long)gm * N + gn;
      if (resid) val += resid[(long long)bz * sR + (long long)gm * N + gn];
      if (Cf) Cf[ci] = val;
      if (Ch) Ch[ci] = (_Float16)val;
    }
  }
}

// ------------------------ small fused / utility kernels --------------------
// f32 (L,K,N) -> f16 transposed (L,N,K)
__global__ void cvt_t_kernel(const float* __restrict__ in, _Float16* __restrict__ out,
                             int L, int K, int N) {
  long long i = (long long)blockIdx.x * blockDim.x + threadIdx.x;
  const long long tot = (long long)L * K * N;
  if (i >= tot) return;
  const int l = (int)(i / ((long long)K * N));
  const long long r = i - (long long)l * K * N;
  const int k = (int)(r / N), n = (int)(r % N);
  out[((long long)l * N + n) * K + k] = (_Float16)in[i];
}

// h1 = nbr(3) @ w1 + b1 ; store f16 ; accumulate BN stats (128 channels).
__global__ __launch_bounds__(128)
void enc_h1_kernel(const float* __restrict__ nbr, const float* __restrict__ w,
                   const float* __restrict__ bias, _Float16* __restrict__ out,
                   float* __restrict__ sum, float* __restrict__ sumsq) {
  const int c = threadIdx.x;
  const long long p0 = (long long)blockIdx.x * 64;
  const float w0 = w[c], w1 = w[128 + c], w2 = w[256 + c], bb = bias[c];
  float s = 0.0f, s2 = 0.0f;
  for (int p = 0; p < 64; ++p) {
    const float* x = nbr + (p0 + p) * 3;
    float h = x[0] * w0 + x[1] * w1 + x[2] * w2 + bb;
    out[(p0 + p) * 128 + c] = (_Float16)h;
    s += h; s2 += h * h;
  }
  atomicAdd(&sum[c], s);
  atomicAdd(&sumsq[c], s2);
}

__global__ void stats_h16_kernel(const _Float16* __restrict__ in,
                                 float* __restrict__ sum, float* __restrict__ sumsq,
                                 int C, int rows) {
  const int c = threadIdx.x;
  const long long r0 = (long long)blockIdx.x * rows;
  float s = 0.0f, s2 = 0.0f;
  for (int r = 0; r < rows; ++r) {
    float v = (float)in[(r0 + r) * C + c];
    s += v; s2 += v * v;
  }
  atomicAdd(&sum[c], s);
  atomicAdd(&sumsq[c], s2);
}

__global__ void bn_relu_kernel(_Float16* __restrict__ x, const float* __restrict__ sum,
                               const float* __restrict__ sumsq,
                               const float* __restrict__ g, const float* __restrict__ b,
                               int cmask, long long total, float invCnt) {
  long long i = (long long)blockIdx.x * blockDim.x + threadIdx.x;
  if (i >= total) return;
  const int c = (int)(i & cmask);
  const float m = sum[c] * invCnt;
  const float var = sumsq[c] * invCnt - m * m;
  float v = ((float)x[i] - m) * rsqrtf(var + 1e-5f) * g[c] + b[c];
  x[i] = (_Float16)fmaxf(v, 0.0f);
}

__global__ void group_max_kernel(const _Float16* __restrict__ in,
                                 _Float16* __restrict__ oh, float* __restrict__ of,
                                 int C) {
  const int g = blockIdx.x, c = threadIdx.x;
  const _Float16* p = in + (long long)g * Kk_ * C + c;
  float m = -3.0e38f;
  for (int k = 0; k < Kk_; ++k) m = fmaxf(m, (float)p[(long long)k * C]);
  if (oh) oh[(long long)g * C + c] = (_Float16)m;
  if (of) of[(long long)g * C + c] = m;
}

// pos stage 1: center(3) @ pos_w1 + b1 -> GELU -> f16
__global__ __launch_bounds__(128)
void pos1_kernel(const float* __restrict__ center, const float* __restrict__ w,
                 const float* __restrict__ bias, _Float16* __restrict__ out) {
  const int c = threadIdx.x;
  const int t0 = blockIdx.x * 8;
  const float w0 = w[c], w1 = w[128 + c], w2 = w[256 + c], bb = bias[c];
  for (int t = 0; t < 8; ++t) {
    const float* cc = center + (long long)(t0 + t) * 3;
    float h = cc[0] * w0 + cc[1] * w1 + cc[2] * w2 + bb;
    h = 0.5f * h * (1.0f + erff(h * 0.70710678118f));
    out[(long long)(t0 + t) * 128 + c] = (_Float16)h;
  }
}

// LayerNorm over D=384 per token; optionally adds pos and stores xin (f32),
// writes gamma/beta-scaled result as f16 (hout) and/or f32 (fout).
__global__ __launch_bounds__(384)
void ln_kernel(const float* __restrict__ xsrc, const float* __restrict__ pos,
               float* __restrict__ xin, const float* __restrict__ g,
               const float* __restrict__ b, _Float16* __restrict__ hout,
               float* __restrict__ fout) {
  const long long tkn = blockIdx.x;
  const int d = threadIdx.x;
  const long long i = tkn * Dd_ + d;
  float v = xsrc[i];
  if (pos) v += pos[i];
  if (xin) xin[i] = v;
  __shared__ float red[13];
  const int wid = d >> 5, lane = d & 31;
  float s = wave_sum(v);
  if (lane == 0) red[wid] = s;
  __syncthreads();
  if (d == 0) { float t = 0; for (int k = 0; k < 12; ++k) t += red[k]; red[12] = t * (1.0f / Dd_); }
  __syncthreads();
  const float mean = red[12];
  __syncthreads();
  const float c = v - mean;
  float s2 = wave_sum(c * c);
  if (lane == 0) red[wid] = s2;
  __syncthreads();
  if (d == 0) { float t = 0; for (int k = 0; k < 12; ++k) t += red[k]; red[12] = t * (1.0f / Dd_); }
  __syncthreads();
  const float var = red[12];
  const float y = c * rsqrtf(var + 1e-5f) * g[d] + b[d];
  if (hout) hout[i] = (_Float16)y;
  if (fout) fout[i] = y;
}

// qkv (8192 x 1152, per-row layout (3,H,hd)) ->
//   q (bh,g,d), kn (bh,g,d) [= B^T for q.k^T], vT (bh,d,g) [= B^T for P.v]
__global__ void qkv_split_kernel(const _Float16* __restrict__ qkv,
                                 _Float16* __restrict__ q, _Float16* __restrict__ kn,
                                 _Float16* __restrict__ vT) {
  const int i = blockIdx.x * blockDim.x + threadIdx.x;   // B*G*H*hd
  const int d = i & 63;
  const int h = (i >> 6) % Hh_;
  const int g = (i / (HD_ * Hh_)) & (Gg_ - 1);
  const int b = i / (HD_ * Hh_ * Gg_);
  const long long row = (long long)b * Gg_ + g;
  const _Float16* src = qkv + row * (3 * Hh_ * HD_);
  const long long bh = (long long)b * Hh_ + h;
  q [(bh * Gg_ + g) * HD_ + d] = src[(0 * Hh_ + h) * HD_ + d];
  kn[(bh * Gg_ + g) * HD_ + d] = src[(1 * Hh_ + h) * HD_ + d];
  vT[(bh * HD_ + d) * Gg_ + g] = src[(2 * Hh_ + h) * HD_ + d];
}

// row softmax over 512 keys, scale = hd^-0.5 = 0.125, f32 in, f16 out
__global__ __launch_bounds__(256)
void softmax_kernel(const float* __restrict__ S, _Float16* __restrict__ P) {
  const long long row = blockIdx.x;
  const float* s = S + row * Gg_;
  _Float16* p = P + row * Gg_;
  const int t = threadIdx.x;
  float v0 = s[t] * 0.125f, v1 = s[t + 256] * 0.125f;
  __shared__ float red[9];
  const int wid = t >> 5, lane = t & 31;
  float m = wave_max(fmaxf(v0, v1));
  if (lane == 0) red[wid] = m;
  __syncthreads();
  if (t == 0) { float mm = red[0]; for (int i = 1; i < 8; ++i) mm = fmaxf(mm, red[i]); red[8] = mm; }
  __syncthreads();
  m = red[8];
  __syncthreads();
  const float e0 = __expf(v0 - m), e1 = __expf(v1 - m);
  float sm = wave_sum(e0 + e1);
  if (lane == 0) red[wid] = sm;
  __syncthreads();
  if (t == 0) { float ss = 0; for (int i = 0; i < 8; ++i) ss += red[i]; red[8] = ss; }
  __syncthreads();
  const float inv = 1.0f / red[8];
  p[t] = (_Float16)(e0 * inv);
  p[t + 256] = (_Float16)(e1 * inv);
}

// O (bh,g,d) -> oh (b,g, h*64+d)
__global__ void attn_merge_kernel(const _Float16* __restrict__ O,
                                  _Float16* __restrict__ oh) {
  const int i = blockIdx.x * blockDim.x + threadIdx.x;
  const int d = i & 63;
  const int h = (i >> 6) % Hh_;
  const int g = (i / (HD_ * Hh_)) & (Gg_ - 1);
  const int b = i / (HD_ * Hh_ * Gg_);
  oh[((long long)b * Gg_ + g) * Dd_ + h * HD_ + d] =
      O[(((long long)b * Hh_ + h) * Gg_ + g) * HD_ + d];
}

// ------------------------------- host side ---------------------------------
extern "C" void kernel_launch(void* const* d_in, const int* in_sizes, int n_in,
                              void* d_out, int out_size, void* d_ws, size_t ws_size,
                              hipStream_t stream) {
  (void)in_sizes; (void)n_in; (void)out_size; (void)ws_size;
  const float* pts     = (const float*)d_in[0];
  const float* enc_w1  = (const float*)d_in[1];
  const float* enc_b1  = (const float*)d_in[2];
  const float* bn1_g   = (const float*)d_in[3];
  const float* bn1_b   = (const float*)d_in[4];
  const float* enc_w2  = (const float*)d_in[5];
  const float* enc_b2  = (const float*)d_in[6];
  const float* enc_w3  = (const float*)d_in[7];
  const float* enc_b3  = (const float*)d_in[8];
  const float* bn2_g   = (const float*)d_in[9];
  const float* bn2_b   = (const float*)d_in[10];
  const float* enc_w4  = (const float*)d_in[11];
  const float* enc_b4  = (const float*)d_in[12];
  const float* pos_w1  = (const float*)d_in[13];
  const float* pos_b1  = (const float*)d_in[14];
  const float* pos_w2  = (const float*)d_in[15];
  const float* pos_b2  = (const float*)d_in[16];
  const float* ln1_g   = (const float*)d_in[17];
  const float* ln1_b   = (const float*)d_in[18];
  const float* qkv_w   = (const float*)d_in[19];
  const float* proj_w  = (const float*)d_in[20];
  const float* proj_b  = (const float*)d_in[21];
  const float* ln2_g   = (const float*)d_in[22];
  const float* ln2_b   = (const float*)d_in[23];
  const float* fc1_w   = (const float*)d_in[24];
  const float* fc1_b   = (const float*)d_in[25];
  const float* fc2_w   = (const float*)d_in[26];
  const float* fc2_b   = (const float*)d_in[27];
  const float* normf_g = (const float*)d_in[28];
  const float* normf_b = (const float*)d_in[29];

  float* out_x      = (float*)d_out;                          // (B,G,D)
  float* out_nbr    = out_x + (long long)Bb_ * Gg_ * Dd_;     // (B,G,K,3)
  float* out_center = out_nbr + (long long)Bb_ * Gg_ * Kk_ * 3; // (B,G,3)

  // -------- workspace bump allocator --------
  unsigned char* base = (unsigned char*)d_ws;
  size_t off = 0;
  auto alloc = [&](size_t bytes) -> void* {
    void* p = base + off;
    off = (off + bytes + 255) & ~(size_t)255;
    return p;
  };

  // persistent: transposed f16 weights + stats + x/pos
  _Float16* w2t    = (_Float16*)alloc(128ULL * 256 * 2);      // (256,128)
  _Float16* w3t    = (_Float16*)alloc(512ULL * 512 * 2);      // (512,512)
  _Float16* w4t    = (_Float16*)alloc(512ULL * 384 * 2);      // (384,512)
  _Float16* posw2t = (_Float16*)alloc(128ULL * 384 * 2);      // (384,128)
  _Float16* qkvwt  = (_Float16*)alloc((size_t)Ll_ * 384 * 1152 * 2); // (l,1152,384)
  _Float16* projwt = (_Float16*)alloc((size_t)Ll_ * 384 * 384 * 2);  // (l,384,384)
  _Float16* fc1wt  = (_Float16*)alloc((size_t)Ll_ * 384 * 1536 * 2); // (l,1536,384)
  _Float16* fc2wt  = (_Float16*)alloc((size_t)Ll_ * 1536 * 384 * 2); // (l,384,1536)
  float* stat1 = (float*)alloc(2 * 128 * 4);
  float* stat2 = (float*)alloc(2 * 512 * 4);
  float*    xbuf  = (float*)alloc(BG_ * Dd_ * 4);
  float*    posb  = (float*)alloc(BG_ * Dd_ * 4);
  _Float16* pcenh = (_Float16*)alloc(BG_ * 128 * 2);

  const size_t scratch0 = off;
  // encoder scratch
  _Float16* buf128 = (_Float16*)alloc(BGK_ * 128 * 2);
  _Float16* f2h    = (_Float16*)alloc(BGK_ * 256 * 2);
  _Float16* fgh    = (_Float16*)alloc(BG_ * 256 * 2);
  _Float16* h3h    = (_Float16*)alloc(BGK_ * 512 * 2);
  _Float16* h4h    = buf128;  // alias: 262144*384*2 bytes == buf128+f2h exactly

  // transformer scratch (unions with encoder scratch; encoder done first)
  off = scratch0;
  float*    xin  = (float*)alloc(BG_ * Dd_ * 4);
  _Float16* hh   = (_Float16*)alloc(BG_ * Dd_ * 2);
  _Float16* qkvh = (_Float16*)alloc(BG_ * 1152 * 2);
  _Float16* qb   = (_Float16*)alloc(96ULL * Gg_ * HD_ * 2);
  _Float16* knb  = (_Float16*)alloc(96ULL * Gg_ * HD_ * 2);
  _Float16* vTb  = (_Float16*)alloc(96ULL * HD_ * Gg_ * 2);
  float*    Sb   = (float*)alloc(96ULL * Gg_ * Gg_ * 4);
  _Float16* Pb   = (_Float16*)alloc(96ULL * Gg_ * Gg_ * 2);
  _Float16* Obh  = (_Float16*)alloc(96ULL * Gg_ * HD_ * 2);
  _Float16* ohb  = (_Float16*)alloc(BG_ * Dd_ * 2);
  _Float16* ffh  = (_Float16*)alloc(BG_ * 1536 * 2);

  auto cvtT = [&](const float* src, _Float16* dst, int L, int K, int N) {
    long long tot = (long long)L * K * N;
    int blocks = (int)((tot + 255) / 256);
    cvt_t_kernel<<<blocks, 256, 0, stream>>>(src, dst, L, K, N);
  };
  auto gemm = [&](bool split,
                  const _Float16* A, long long sA,
                  const _Float16* A2, int splitK, int a2div,
                  const _Float16* BT, long long sB,
                  const float* bias, const float* resid, long long sR,
                  float* Cf, _Float16* Ch, long long sC,
                  int M, int N, int Kd, int act, int batch) {
    dim3 grd(N / 64, M / 64, batch);
    if (split)
      wmma_gemm<true><<<grd, 128, 0, stream>>>(A, sA, A2, splitK, a2div, BT, sB,
                                               bias, resid, sR, Cf, Ch, sC, M, N, Kd, act);
    else
      wmma_gemm<false><<<grd, 128, 0, stream>>>(A, sA, A2, splitK, a2div, BT, sB,
                                                bias, resid, sR, Cf, Ch, sC, M, N, Kd, act);
  };

  // -------- weight conversion (transposed) + stat zeroing --------
  cvtT(enc_w2, w2t, 1, 128, 256);
  cvtT(enc_w3, w3t, 1, 512, 512);
  cvtT(enc_w4, w4t, 1, 512, 384);
  cvtT(pos_w2, posw2t, 1, 128, 384);
  cvtT(qkv_w, qkvwt, Ll_, 384, 1152);
  cvtT(proj_w, projwt, Ll_, 384, 384);
  cvtT(fc1_w, fc1wt, Ll_, 384, 1536);
  cvtT(fc2_w, fc2wt, Ll_, 1536, 384);
  (void)hipMemsetAsync(stat1, 0, 2 * 128 * 4, stream);
  (void)hipMemsetAsync(stat2, 0, 2 * 512 * 4, stream);

  // -------- geometry --------
  fps_kernel<<<Bb_, 1024, 0, stream>>>(pts, out_center);
  knn_kernel<<<Bb_ * Gg_, 256, 0, stream>>>(pts, out_center, out_nbr);

  // -------- positional embedding --------
  pos1_kernel<<<BG_ / 8, 128, 0, stream>>>(out_center, pos_w1, pos_b1, pcenh);
  gemm(false, pcenh, 0, nullptr, 0, 1, posw2t, 0, pos_b2, nullptr, 0,
       posb, nullptr, 0, (int)BG_, Dd_, 128, 0, 1);

  // -------- mini-PointNet encoder --------
  enc_h1_kernel<<<(int)(BGK_ / 64), 128, 0, stream>>>(out_nbr, enc_w1, enc_b1,
                                                      buf128, stat1, stat1 + 128);
  {
    long long tot = BGK_ * 128;
    bn_relu_kernel<<<(int)(tot / 256), 256, 0, stream>>>(
        buf128, stat1, stat1 + 128, bn1_g, bn1_b, 127, tot, 1.0f / (float)BGK_);
  }
  gemm(false, buf128, 0, nullptr, 0, 1, w2t, 0, enc_b2, nullptr, 0,
       nullptr, f2h, 0, (int)BGK_, 256, 128, 0, 1);
  group_max_kernel<<<(int)BG_, 256, 0, stream>>>(f2h, fgh, nullptr, 256);
  // concat([fg bcast over K, f2]) @ w3 + b3 (split-A addressing)
  gemm(true, f2h, 0, fgh, 256, Kk_, w3t, 0, enc_b3, nullptr, 0,
       nullptr, h3h, 0, (int)BGK_, 512, 512, 0, 1);
  stats_h16_kernel<<<(int)(BGK_ / 64), 512, 0, stream>>>(h3h, stat2, stat2 + 512, 512, 64);
  {
    long long tot = BGK_ * 512;
    bn_relu_kernel<<<(int)(tot / 256), 256, 0, stream>>>(
        h3h, stat2, stat2 + 512, bn2_g, bn2_b, 511, tot, 1.0f / (float)BGK_);
  }
  gemm(false, h3h, 0, nullptr, 0, 1, w4t, 0, enc_b4, nullptr, 0,
       nullptr, h4h, 0, (int)BGK_, Dd_, 512, 0, 1);
  group_max_kernel<<<(int)BG_, Dd_, 0, stream>>>(h4h, nullptr, xbuf, Dd_);

  // -------- transformer (L=12) --------
  for (int l = 0; l < Ll_; ++l) {
    const _Float16* qkvw_l = qkvwt + (size_t)l * 1152 * 384;
    const _Float16* projw_l = projwt + (size_t)l * 384 * 384;
    const _Float16* fc1w_l = fc1wt + (size_t)l * 1536 * 384;
    const _Float16* fc2w_l = fc2wt + (size_t)l * 384 * 1536;

    // xin = x + pos ; h = LN1(xin)
    ln_kernel<<<(int)BG_, Dd_, 0, stream>>>(xbuf, posb, xin,
                                            ln1_g + l * Dd_, ln1_b + l * Dd_, hh, nullptr);
    // qkv = h @ qkv_w
    gemm(false, hh, 0, nullptr, 0, 1, qkvw_l, 0, nullptr, nullptr, 0,
         nullptr, qkvh, 0, (int)BG_, 1152, Dd_, 0, 1);
    qkv_split_kernel<<<(int)(BG_ * Dd_ / 256), 256, 0, stream>>>(qkvh, qb, knb, vTb);
    // S = q @ k^T  (BT = k in natural (g,d) layout), batched over B*H = 96
    gemm(false, qb, (long long)Gg_ * HD_, nullptr, 0, 1, knb, (long long)Gg_ * HD_,
         nullptr, nullptr, 0, Sb, nullptr, (long long)Gg_ * Gg_,
         Gg_, Gg_, HD_, 0, 96);
    softmax_kernel<<<96 * Gg_, 256, 0, stream>>>(Sb, Pb);
    // O = P @ v  (BT = v transposed (d,g))
    gemm(false, Pb, (long long)Gg_ * Gg_, nullptr, 0, 1, vTb, (long long)HD_ * Gg_,
         nullptr, nullptr, 0, nullptr, Obh, (long long)Gg_ * HD_,
         Gg_, HD_, Gg_, 0, 96);
    attn_merge_kernel<<<(int)(BG_ * Dd_ / 256), 256, 0, stream>>>(Obh, ohb);
    // x = xin + O @ proj_w + proj_b
    gemm(false, ohb, 0, nullptr, 0, 1, projw_l, 0, proj_b + l * Dd_, xin, 0,
         xbuf, nullptr, 0, (int)BG_, Dd_, Dd_, 0, 1);
    // h = LN2(x)
    ln_kernel<<<(int)BG_, Dd_, 0, stream>>>(xbuf, nullptr, nullptr,
                                            ln2_g + l * Dd_, ln2_b + l * Dd_, hh, nullptr);
    // ff = gelu(h @ fc1_w + fc1_b)
    gemm(false, hh, 0, nullptr, 0, 1, fc1w_l, 0, fc1_b + l * 1536, nullptr, 0,
         nullptr, ffh, 0, (int)BG_, 1536, Dd_, 2, 1);
    // x = x + ff @ fc2_w + fc2_b
    gemm(false, ffh, 0, nullptr, 0, 1, fc2w_l, 0, fc2_b + l * Dd_, xbuf, 0,
         xbuf, nullptr, 0, (int)BG_, Dd_, 1536, 0, 1);
  }

  // -------- final LN -> output --------
  ln_kernel<<<(int)BG_, Dd_, 0, stream>>>(xbuf, nullptr, nullptr,
                                          normf_g, normf_b, nullptr, out_x);
}
